// Qwen34BAttentionKernel_87273735455081
// MI455X (gfx1250) — matrix operand order repaired and checked by reading, not compiled, not run on hardware
//
#include <hip/hip_runtime.h>
#include <hip/hip_bf16.h>

// ---------------------------------------------------------------------------
// Qwen3-4B-style attention block for MI455X (gfx1250, wave32, WMMA).
//   B=4, S=1024, D=2048, NH=16, HD=128, SCALE = 128^-0.5
// Pipeline:
//   1) gemm_bf16_wmma<0>: qkv = query @ Wqkv^T + bqkv  -> scatter to bf16
//      per-head Q (scaled), K row-major, V (odd permutation) transposed.
//   2) attn_kernel: causal flash attention, bf16 WMMA, fp32 accum,
//      double-buffered GLOBAL_LOAD_ASYNC_TO_LDS staging (ASYNCcnt).
//   3) gemm_bf16_wmma<1>: out = O @ Wout^T + bout (fp32).
// ---------------------------------------------------------------------------

typedef float          f32x4  __attribute__((ext_vector_type(4)));
typedef float          f32x8  __attribute__((ext_vector_type(8)));
typedef unsigned short u16x4  __attribute__((ext_vector_type(4)));
typedef unsigned short u16x8  __attribute__((ext_vector_type(8)));
typedef unsigned short u16x16 __attribute__((ext_vector_type(16)));
typedef __bf16         bf16x16 __attribute__((ext_vector_type(16)));

#define NB   4
#define SEQ  1024
#define DIM  2048
#define NH   16
#define HD   128
#define ATT_SCALE 0.08838834764831845f   // 128^-0.5

// fp32 -> bf16 via hardware convert
__device__ __forceinline__ unsigned short f2bf(float f) {
    return __builtin_bit_cast(unsigned short, (__bf16)f);
}

// Assemble a 16-element bf16 fragment from two contiguous 8x16b loads.
__device__ __forceinline__ bf16x16 mk_frag(u16x8 lo, u16x8 hi) {
    u16x16 t;
#pragma unroll
    for (int i = 0; i < 8; ++i) { t[i] = lo[i]; t[i + 8] = hi[i]; }
    return __builtin_bit_cast(bf16x16, t);
}

__device__ __forceinline__ f32x8 wmma_bf16(bf16x16 a, bf16x16 b, f32x8 c) {
    return __builtin_amdgcn_wmma_f32_16x16x32_bf16(
        false, a, false, b, (short)0, c, false, false);
}

// CDNA5 async copy: global -> LDS, 16 bytes per lane, tracked by ASYNCcnt.
// lds_addr = low 32 bits of a flat shared pointer (LDS byte offset),
// goff     = per-lane signed byte offset, sbase = uniform 64-bit base.
__device__ __forceinline__ void async_b128(unsigned lds_addr, int goff,
                                           const unsigned short* sbase) {
    asm volatile("global_load_async_to_lds_b128 %0, %1, %2"
                 :: "v"(lds_addr), "v"(goff), "s"(sbase)
                 : "memory");
}
__device__ __forceinline__ void wait_async0() {
    asm volatile("s_wait_asynccnt 0x0" ::: "memory");
}

// ---------------------------------------------------------------------------
// GEMM: Y(M x N) = A(M x K) @ W(N x K)^T + bias, bf16 WMMA, fp32 accumulate.
// Block = 256 threads (8 waves). Tile: 128 rows x 128 cols. K-step = 32.
// Each wave: 1 A-frag, 8 B-frags, 8 WMMAs per K-step. Register-prefetch
// pipeline keeps loadcnt waits behind a full WMMA phase.
// MODE 0: QKV epilogue -> scatter bf16 Qh/Kh (row-major) + Vh (transposed).
// MODE 1: plain fp32 output (row-major M x N).
// ---------------------------------------------------------------------------
template <int MODE>
__global__ __launch_bounds__(256)
void gemm_bf16_wmma(const float* __restrict__ A, const float* __restrict__ W,
                    const float* __restrict__ bias, int K, int N,
                    unsigned short* __restrict__ Qh,
                    unsigned short* __restrict__ Kh,
                    unsigned short* __restrict__ Vh,
                    float* __restrict__ Out) {
    __shared__ unsigned short Xs[128][32];
    __shared__ unsigned short Ws[128][32];

    const int t    = threadIdx.x;
    const int wave = t >> 5;
    const int l    = t & 31;
    const int hi   = l >> 4;    // lane half
    const int ln   = l & 15;
    const int m0   = blockIdx.y * 128;
    const int n0   = blockIdx.x * 128;

    f32x8 acc[8];
#pragma unroll
    for (int ct = 0; ct < 8; ++ct)
#pragma unroll
        for (int e = 0; e < 8; ++e) acc[ct][e] = 0.0f;

    const int r8 = t >> 3;          // 0..31
    const int c4 = (t & 7) * 4;     // 0..28

    // ---- register prefetch staging ----
    f32x4 xr[4], wr[4];
#define GEMM_LOAD(k0_)                                                         \
    {                                                                          \
        _Pragma("unroll")                                                      \
        for (int i = 0; i < 4; ++i)                                            \
            xr[i] = *(const f32x4*)(A + (size_t)(m0 + r8 + 32 * i) * K +       \
                                    (k0_) + c4);                               \
        _Pragma("unroll")                                                      \
        for (int i = 0; i < 4; ++i)                                            \
            wr[i] = *(const f32x4*)(W + (size_t)(n0 + r8 + 32 * i) * K +       \
                                    (k0_) + c4);                               \
    }

    GEMM_LOAD(0);

    for (int k0 = 0; k0 < K; k0 += 32) {
        __syncthreads();               // prior iteration's frag reads done
        // convert + store staged registers into LDS
#pragma unroll
        for (int i = 0; i < 4; ++i) {
            u16x4 px, pw;
#pragma unroll
            for (int e = 0; e < 4; ++e) { px[e] = f2bf(xr[i][e]); pw[e] = f2bf(wr[i][e]); }
            *(u16x4*)&Xs[r8 + 32 * i][c4] = px;
            *(u16x4*)&Ws[r8 + 32 * i][c4] = pw;
        }
        __syncthreads();
        // issue next stage's global loads (waited at next iteration's store)
        if (k0 + 32 < K) GEMM_LOAD(k0 + 32);

        // A fragment: 16x32, lane half kb8 split (K 0..7/16..23 vs 8..15/24..31)
        const int kb8 = hi * 8;
        const unsigned short* ap = &Xs[wave * 16 + ln][kb8];
        const bf16x16 af = mk_frag(*(const u16x8*)ap, *(const u16x8*)(ap + 16));
#pragma unroll
        for (int ct = 0; ct < 8; ++ct) {
            // B fragment: 32x16, lane half kb8 split (K 0..7/16..23 vs 8..15/24..31)
            const unsigned short* bp = &Ws[ct * 16 + ln][hi * 8];
            const bf16x16 bfr = mk_frag(*(const u16x8*)bp, *(const u16x8*)(bp + 16));
            acc[ct] = wmma_bf16(af, bfr, acc[ct]);
        }
    }
#undef GEMM_LOAD

    // Epilogue. C/D layout: lane covers n = ln, rows m = r + 8*hi.
#pragma unroll
    for (int ct = 0; ct < 8; ++ct) {
#pragma unroll
        for (int r = 0; r < 8; ++r) {
            const int m = m0 + wave * 16 + hi * 8 + r;
            const int n = n0 + ct * 16 + ln;
            const float val = acc[ct][r] + bias[n];
            if (MODE == 0) {
                const int b = m >> 10, s = m & 1023;
                if (n < DIM) {                       // Q, pre-scaled
                    Qh[(((size_t)b * NH + (n >> 7)) * SEQ + s) * HD + (n & 127)] =
                        f2bf(val * ATT_SCALE);
                } else if (n < 2 * DIM) {            // K, row-major [s][d]
                    const int c = n - DIM;
                    Kh[(((size_t)b * NH + (c >> 7)) * SEQ + s) * HD + (c & 127)] =
                        f2bf(val);
                } else {                             // V permuted + TRANSPOSED
                    const int c = n - 2 * DIM;
                    // v.reshape(b,s,HD,NH).transpose(0,2,1,3).reshape(b,s,D)
                    //   .reshape(b,s,NH,HD).transpose(0,2,1,3)
                    const int F  = (c >> 4) * (SEQ * NH) + s * NH + (c & 15);
                    const int s2 = F >> 11, c2 = F & 2047;
                    // layout: [b][h][d][s]  (d-major for b128 attention staging)
                    Vh[(((size_t)b * NH + (c2 >> 7)) * HD + (c2 & 127)) * SEQ + s2] =
                        f2bf(val);
                }
            } else {
                Out[(size_t)m * N + n] = val;
            }
        }
    }
}

// ---------------------------------------------------------------------------
// Causal flash attention. Grid (S/64, B*NH); block = 128 threads = 4 waves.
// Each wave owns 16 query rows; the block cooperates on 32-key steps.
// K/V staging uses double-buffered GLOBAL_LOAD_ASYNC_TO_LDS_B128: stage js+1
// is issued right after the barrier of stage js and completes while the wave
// runs 16 WMMAs + softmax on stage js.
// ---------------------------------------------------------------------------
__global__ __launch_bounds__(128)
void attn_kernel(const unsigned short* __restrict__ Qh,
                 const unsigned short* __restrict__ Kh,
                 const unsigned short* __restrict__ Vh,
                 float* __restrict__ O) {
    __shared__ unsigned short Kb[2][32][128];   // K block, row-major [key][d]
    __shared__ unsigned short Vt[2][128][32];   // V block, transposed [d][key]
    __shared__ unsigned short Pl[4][16][32];    // per-wave P tile (bf16)

    const int t    = threadIdx.x;
    const int wave = t >> 5;
    const int l    = t & 31;
    const int hi   = l >> 4;
    const int ln   = l & 15;
    const int kb8  = hi * 8;
    const int kb16 = hi * 16;
    const int bh   = blockIdx.y;                  // b*NH + h
    const int qb   = blockIdx.x * 64 + wave * 16; // wave's first query row
    const size_t base  = (size_t)bh * SEQ * HD;   // Q/K row-major base
    const size_t baseT = (size_t)bh * HD * SEQ;   // V transposed base
    const unsigned short* kGlob = Kh + base;      // uniform 64-bit bases
    const unsigned short* vGlob = Vh + baseT;

    // Resident Q fragments: 16 rows x 128 head-dim = 4 A-frags of 16x32
    bf16x16 qf[4];
    {
        const size_t qoff = base + (size_t)(qb + ln) * HD;
#pragma unroll
        for (int kk = 0; kk < 4; ++kk) {
            const unsigned short* p = Qh + qoff + kk * 32 + kb8;
            qf[kk] = mk_frag(*(const u16x8*)p, *(const u16x8*)(p + 16));
        }
    }

    f32x8 o[8];
    float m_i[8], l_i[8];
#pragma unroll
    for (int nt = 0; nt < 8; ++nt)
#pragma unroll
        for (int e = 0; e < 8; ++e) o[nt][e] = 0.0f;
#pragma unroll
    for (int r = 0; r < 8; ++r) { m_i[r] = -3e38f; l_i[r] = 0.0f; }

    const int jsteps = 2 * (blockIdx.x + 1);     // keys up to block_end

    // async staging geometry (128 threads)
    const int krow  = t >> 3;          // 0..15 : two K rows per thread
    const int kcol8 = (t & 7) * 16;    // 0..112 step16
    const int vd    = t;               // 0..127 : one V (transposed) row each

    // Issue one 32-key stage (8 async b128 per thread) into buffer `buf`.
    auto issue_stage = [&](int jb_, int buf) {
#pragma unroll
        for (int g = 0; g < 2; ++g) {
            const int row  = krow + 16 * g;
            const int goff = ((jb_ + row) * HD + kcol8) * 2;   // bytes
            async_b128((unsigned)(size_t)&Kb[buf][row][kcol8],     goff,      kGlob);
            async_b128((unsigned)(size_t)&Kb[buf][row][kcol8 + 8], goff + 16, kGlob);
        }
#pragma unroll
        for (int g = 0; g < 4; ++g) {
            const int goff = (vd * SEQ + jb_ + 8 * g) * 2;     // bytes
            async_b128((unsigned)(size_t)&Vt[buf][vd][8 * g], goff, vGlob);
        }
    };

    issue_stage(0, 0);

    for (int js = 0; js < jsteps; ++js) {
        const int jb  = js * 32;
        const int buf = js & 1;
        wait_async0();                 // this wave's stage js landed in LDS
        __syncthreads();               // all waves: js visible, js-1 reads done
        if (js + 1 < jsteps)           // overlap next stage with compute below
            issue_stage(jb + 32, buf ^ 1);

        // S = Q @ K^T : two 16-key column tiles, K frags from LDS
        f32x8 sc[2];
#pragma unroll
        for (int tt = 0; tt < 2; ++tt) {
#pragma unroll
            for (int e = 0; e < 8; ++e) sc[tt][e] = 0.0f;
#pragma unroll
            for (int kk = 0; kk < 4; ++kk) {
                const unsigned short* p = &Kb[buf][tt * 16 + ln][kk * 32 + kb8];
                const bf16x16 bk = mk_frag(*(const u16x8*)p, *(const u16x8*)(p + 16));
                sc[tt] = wmma_bf16(qf[kk], bk, sc[tt]);
            }
        }

        // Causal mask + online softmax (row reductions across 16 lanes)
#pragma unroll
        for (int r = 0; r < 8; ++r) {
            const int qrow = qb + hi * 8 + r;
            const float s0 = (jb + ln > qrow)      ? -1e30f : sc[0][r];
            const float s1 = (jb + 16 + ln > qrow) ? -1e30f : sc[1][r];
            float rm = fmaxf(s0, s1);
            rm = fmaxf(rm, __shfl_xor(rm, 1, 32));
            rm = fmaxf(rm, __shfl_xor(rm, 2, 32));
            rm = fmaxf(rm, __shfl_xor(rm, 4, 32));
            rm = fmaxf(rm, __shfl_xor(rm, 8, 32));
            const float nm    = fmaxf(m_i[r], rm);
            const float alpha = __expf(m_i[r] - nm);
            m_i[r] = nm;
            const float p0 = __expf(s0 - nm);
            const float p1 = __expf(s1 - nm);
            float rs = p0 + p1;
            rs += __shfl_xor(rs, 1, 32);
            rs += __shfl_xor(rs, 2, 32);
            rs += __shfl_xor(rs, 4, 32);
            rs += __shfl_xor(rs, 8, 32);
            l_i[r] = l_i[r] * alpha + rs;
#pragma unroll
            for (int nt = 0; nt < 8; ++nt) o[nt][r] *= alpha;
            Pl[wave][hi * 8 + r][ln]      = f2bf(p0);
            Pl[wave][hi * 8 + r][16 + ln] = f2bf(p1);
        }
        // within-wave LDS RAW: DS is in-order per wave; fence compiler + ctr
        asm volatile("s_wait_dscnt 0x0" ::: "memory");

        // P as A-fragment (16x32) from per-wave LDS
        bf16x16 pa;
        {
            const unsigned short* p = &Pl[wave][ln][kb8];
            pa = mk_frag(*(const u16x8*)p, *(const u16x8*)(p + 16));
        }
        // O += P @ V : 8 n-tiles of 16 head-dim columns
#pragma unroll
        for (int nt = 0; nt < 8; ++nt) {
            const unsigned short* p = &Vt[buf][nt * 16 + ln][kb8];
            const bf16x16 bv = mk_frag(*(const u16x8*)p, *(const u16x8*)(p + 16));
            o[nt] = wmma_bf16(pa, bv, o[nt]);
        }
    }

    // Epilogue: write (b, s, h*HD + d) fp32 for the out-projection GEMM
    const int b = bh >> 4, h = bh & 15;
#pragma unroll
    for (int nt = 0; nt < 8; ++nt) {
#pragma unroll
        for (int r = 0; r < 8; ++r) {
            const int row = qb + hi * 8 + r;
            const int d   = nt * 16 + ln;
            O[((size_t)(b * SEQ + row)) * DIM + h * HD + d] = o[nt][r] / l_i[r];
        }
    }
}

// ---------------------------------------------------------------------------
extern "C" void kernel_launch(void* const* d_in, const int* in_sizes, int n_in,
                              void* d_out, int out_size, void* d_ws, size_t ws_size,
                              hipStream_t stream) {
    (void)in_sizes; (void)n_in; (void)out_size;
    const float* query = (const float*)d_in[0];
    // d_in[1] (key) / d_in[2] (value) are ignored by the reference module.
    const float* Wqkv  = (const float*)d_in[3];
    const float* bqkv  = (const float*)d_in[4];
    const float* Wout  = (const float*)d_in[5];
    const float* bout  = (const float*)d_in[6];
    float*       out   = (float*)d_out;

    // Workspace layout (bytes):
    const size_t bh_elems = (size_t)NB * NH * SEQ * HD;   // 8,388,608
    const size_t qh_off = 0;
    const size_t kh_off = qh_off + bh_elems * 2;
    const size_t vh_off = kh_off + bh_elems * 2;
    const size_t o_off  = vh_off + bh_elems * 2;
    const size_t need   = o_off + (size_t)NB * SEQ * DIM * 4;
    if (ws_size < need) return;   // deterministic no-op if scratch too small

    unsigned short* Qh = (unsigned short*)((char*)d_ws + qh_off);
    unsigned short* Kh = (unsigned short*)((char*)d_ws + kh_off);
    unsigned short* Vh = (unsigned short*)((char*)d_ws + vh_off);
    float*          Osc = (float*)((char*)d_ws + o_off);

    // 1) QKV projection: M=4096, N=6144, K=2048 -> scatter bf16 heads
    {
        dim3 grid(6144 / 128, 4096 / 128);
        gemm_bf16_wmma<0><<<grid, 256, 0, stream>>>(
            query, Wqkv, bqkv, DIM, 3 * DIM, Qh, Kh, Vh, nullptr);
    }
    // 2) Causal flash attention
    {
        dim3 grid(SEQ / 64, NB * NH);
        attn_kernel<<<grid, 128, 0, stream>>>(Qh, Kh, Vh, Osc);
    }
    // 3) Output projection: M=4096, N=2048, K=2048 -> d_out fp32
    {
        dim3 grid(2048 / 128, 4096 / 128);
        gemm_bf16_wmma<1><<<grid, 256, 0, stream>>>(
            Osc, Wout, bout, DIM, DIM, nullptr, nullptr, nullptr, out);
    }
}
